// HilbertAttentionTritonWrapper_46188078301814
// MI455X (gfx1250) — matrix operand order repaired and checked by reading, not compiled, not run on hardware
//
#include <hip/hip_runtime.h>
#include <hip/hip_bf16.h>

// ---------------------------------------------------------------------------
// HilbertAttention for MI455X (gfx1250, wave32, WMMA).
// Pipeline: hilbert perm -> head-mean -> reduced QKV GEMM (wmma f16) ->
//           segmented flash attention (wmma f16, async-LDS staged K, V^T) ->
//           output GEMM (wmma f16) -> f32 result.
// ---------------------------------------------------------------------------

typedef __attribute__((ext_vector_type(16))) _Float16 v16h;
typedef __attribute__((ext_vector_type(8)))  _Float16 v8h;
typedef __attribute__((ext_vector_type(8)))  float    v8f;

#define B_    2
#define M_    8192
#define H_    8
#define D_    64
#define HD_   512
#define SEG_  512
#define NSEG_ 16
#define ROWS_ 16384   // B_*M_
#define F3_   1536    // 3*HD_

__device__ __forceinline__ v8f wmma16(v16h a, v16h b, v8f c) {
  // D = A(16x32 f16) * B(32x16 f16) + C(16x16 f32)
  return __builtin_amdgcn_wmma_f32_16x16x32_f16(false, a, false, b, (short)0, c,
                                                false, false);
}

// A-operand load (16-bit A 16x32 layout): lane row = lane%16, hi = lane/16.
// Elements 0..15 -> K = kb + hi*16 + e (contiguous).
// Caller passes p = rowBase + kb + hi*16.
__device__ __forceinline__ v16h load_a16(const _Float16* p) {
  v8h lo = *(const v8h*)p;
  v8h hh = *(const v8h*)(p + 8);
  v16h r;
#pragma unroll
  for (int i = 0; i < 8; ++i) { r[i] = lo[i]; r[8 + i] = hh[i]; }
  return r;
}

// ---------------------------------------------------------------------------
// Hilbert curve: per-index distance then rank (argsort of distinct keys).
// ---------------------------------------------------------------------------
__global__ void hilbert_d_kernel(int* __restrict__ dvals) {
  int i = blockIdx.x * blockDim.x + threadIdx.x;
  if (i >= M_) return;
  int x = i & 127, y = i >> 7, d = 0;
  for (int s = 64; s > 0; s >>= 1) {
    int rx = (x & s) ? 1 : 0;
    int ry = (y & s) ? 1 : 0;
    d += s * s * ((3 * rx) ^ ry);
    if (ry == 0) {
      if (rx == 1) { x = s - 1 - x; y = s - 1 - y; }
      int t = x; x = y; y = t;
    }
  }
  dvals[i] = d;
}

__global__ __launch_bounds__(256) void hilbert_rank_kernel(
    const int* __restrict__ dvals, int* __restrict__ perm, int* __restrict__ inv) {
  __shared__ int dsh[M_];
  for (int j = threadIdx.x; j < M_; j += blockDim.x) dsh[j] = dvals[j];
  __syncthreads();
  int i = blockIdx.x * blockDim.x + threadIdx.x;
  if (i >= M_) return;
  int di = dsh[i], r = 0;
  for (int j = 0; j < M_; ++j) r += (dsh[j] < di) ? 1 : 0;
  perm[r] = i;   // perm = argsort(d)
  inv[i]  = r;   // inv  = argsort(perm)
}

// ---------------------------------------------------------------------------
// x = mean over heads of q, converted to f16.  x_h: (ROWS_, 64)
// ---------------------------------------------------------------------------
__global__ void mean_kernel(const float* __restrict__ q, _Float16* __restrict__ x_h) {
  int idx = blockIdx.x * blockDim.x + threadIdx.x;   // ROWS_*64 threads
  int r = idx >> 6, d = idx & 63;
  const float* qp = q + (size_t)r * HD_ + d;
  float s = 0.f;
#pragma unroll
  for (int h = 0; h < H_; ++h) s += qp[h * D_];
  x_h[idx] = (_Float16)(s * 0.125f);
}

// wr[f][d] = sum_h w_qkv[f][h*64+d]  (exploits the tile(x, H) structure)
__global__ void wr_kernel(const float* __restrict__ w_qkv, _Float16* __restrict__ wr_h) {
  int idx = blockIdx.x * blockDim.x + threadIdx.x;   // 1536*64 threads
  int f = idx >> 6, d = idx & 63;
  const float* wp = w_qkv + (size_t)f * HD_ + d;
  float s = 0.f;
#pragma unroll
  for (int h = 0; h < H_; ++h) s += wp[h * D_];
  wr_h[idx] = (_Float16)s;
}

__global__ void wout_cvt_kernel(const float* __restrict__ w_out, _Float16* __restrict__ wout_h) {
  int idx = blockIdx.x * blockDim.x + threadIdx.x;   // 512*512 threads
  wout_h[idx] = (_Float16)w_out[idx];
}

// ---------------------------------------------------------------------------
// GEMM1: qkvP[b, inv[m], f] = sum_d x_h[b,m,d] * wr_h[f,d]   (K = 64)
// wave = 16 rows x 64 cols ; grid = 1024 row-blocks x 24 col-blocks = 24576 waves
// ---------------------------------------------------------------------------
__global__ __launch_bounds__(256) void qkv_gemm_kernel(
    const _Float16* __restrict__ x_h, const _Float16* __restrict__ wr_h,
    const int* __restrict__ inv, _Float16* __restrict__ qkvP) {
  const int wid = blockIdx.x * 8 + (threadIdx.x >> 5);
  const int rb = wid / 24, nb = wid % 24;
  const int lane = threadIdx.x & 31, lo16 = lane & 15, hi = lane >> 4;

  const _Float16* arow = x_h + (size_t)(rb * 16 + lo16) * 64;
  v16h a0 = load_a16(arow + 0  + hi * 16);
  v16h a1 = load_a16(arow + 32 + hi * 16);

  v8f acc[4];
#pragma unroll
  for (int n = 0; n < 4; ++n)
#pragma unroll
    for (int v = 0; v < 8; ++v) acc[n][v] = 0.f;

#pragma unroll
  for (int n = 0; n < 4; ++n) {
    const _Float16* brow = wr_h + (size_t)(nb * 64 + n * 16 + lo16) * 64;
    acc[n] = wmma16(a0, *(const v16h*)(brow + 0  + hi * 16), acc[n]);
    acc[n] = wmma16(a1, *(const v16h*)(brow + 32 + hi * 16), acc[n]);
  }

#pragma unroll
  for (int v = 0; v < 8; ++v) {
    int r = rb * 16 + v + 8 * hi;                 // global source row
    int bb = r >> 13, mrow = r & (M_ - 1);
    size_t orow = (size_t)bb * M_ + inv[mrow];    // scatter into Hilbert order
#pragma unroll
    for (int n = 0; n < 4; ++n)
      qkvP[orow * F3_ + nb * 64 + n * 16 + lo16] = (_Float16)acc[n][v];
  }
}

// ---------------------------------------------------------------------------
// Flash attention per (b, seg, head, quarter): 8 waves x 16 Q rows = 128 rows.
// K (512x64) staged with async global->LDS DMA (ASYNCcnt); V^T (64x512)
// transposed through VGPRs; per-wave P scratch for C-layout -> A-layout.
// ---------------------------------------------------------------------------
__global__ __launch_bounds__(256) void attn_kernel(
    const _Float16* __restrict__ qkvP, const int* __restrict__ perm,
    _Float16* __restrict__ attn_h) {
  __shared__ __attribute__((aligned(32))) _Float16 kbuf[SEG_ * D_];   // 64 KB
  __shared__ __attribute__((aligned(32))) _Float16 vtbuf[D_ * SEG_];  // 64 KB
  __shared__ __attribute__((aligned(32))) _Float16 pbuf[8 * 16 * 32]; // 8 KB

  const int bnh = blockIdx.x >> 2;
  const int sub = blockIdx.x & 3;
  const int b   = bnh / (NSEG_ * H_);
  const int rem = bnh % (NSEG_ * H_);
  const int seg = rem / H_;
  const int h   = rem % H_;
  const size_t seg_row0 = (size_t)b * M_ + (size_t)seg * SEG_;

  // K tile: async DMA global -> LDS, 16B per lane per issue (no VGPR round-trip).
  // LDS dest address = low 32 bits of the flat pointer (aperture rule).
  for (int i = threadIdx.x; i < SEG_ * D_ / 8; i += 256) {
    int t = i >> 3, dg = (i & 7) * 8;
    const _Float16* gsrc = qkvP + (seg_row0 + t) * F3_ + HD_ + h * 64 + dg;
    unsigned ldst = (unsigned)(size_t)&kbuf[t * D_ + dg];
    asm volatile("global_load_async_to_lds_b128 %0, %1, off"
                 :: "v"(ldst), "v"((unsigned long long)(size_t)gsrc)
                 : "memory");
  }

  // V tile: load rows, store transposed into vtbuf.
  for (int i = threadIdx.x; i < SEG_ * D_ / 8; i += 256) {
    int t = i >> 3, dg = (i & 7) * 8;
    v8h vv = *(const v8h*)(qkvP + (seg_row0 + t) * F3_ + 2 * HD_ + h * 64 + dg);
#pragma unroll
    for (int e = 0; e < 8; ++e) vtbuf[(dg + e) * SEG_ + t] = vv[e];
  }

  asm volatile("s_wait_asynccnt 0" ::: "memory");   // our async K copies done
  __syncthreads();                                  // all waves' staging visible

  const int wid  = threadIdx.x >> 5;
  const int lane = threadIdx.x & 31;
  const int lo16 = lane & 15;
  const int hi   = lane >> 4;

  const int qrow = sub * 128 + wid * 16;   // row offset within segment
  const _Float16* qptr = qkvP + (seg_row0 + qrow + lo16) * F3_ + h * 64;
  v16h aq0 = load_a16(qptr + 0  + hi * 16);
  v16h aq1 = load_a16(qptr + 32 + hi * 16);

  v8f O0, O1, O2, O3;
  float m8[8], l8[8];
#pragma unroll
  for (int v = 0; v < 8; ++v) {
    O0[v] = O1[v] = O2[v] = O3[v] = 0.f;
    m8[v] = -3.0e38f; l8[v] = 0.f;
  }
  const float SC = 0.125f * 1.44269504f;   // D^-0.5 * log2(e)

  for (int t0 = 0; t0 < SEG_; t0 += 32) {
    v8f S0, S1;
#pragma unroll
    for (int v = 0; v < 8; ++v) { S0[v] = 0.f; S1[v] = 0.f; }

    // S = Q * K^T  (B operand = contiguous rows of K at kb + hi*16)
    const _Float16* k0 = &kbuf[(t0      + lo16) * D_];
    const _Float16* k1 = &kbuf[(t0 + 16 + lo16) * D_];
    S0 = wmma16(aq0, *(const v16h*)(k0 + 0  + hi * 16), S0);
    S0 = wmma16(aq1, *(const v16h*)(k0 + 32 + hi * 16), S0);
    S1 = wmma16(aq0, *(const v16h*)(k1 + 0  + hi * 16), S1);
    S1 = wmma16(aq1, *(const v16h*)(k1 + 32 + hi * 16), S1);

    // Online softmax (row = v + 8*hi lives in 16-lane half; reduce with shfl_xor).
#pragma unroll
    for (int v = 0; v < 8; ++v) {
      S0[v] *= SC; S1[v] *= SC;
      float t = fmaxf(S0[v], S1[v]);
      t = fmaxf(t, __shfl_xor(t, 1, 32));
      t = fmaxf(t, __shfl_xor(t, 2, 32));
      t = fmaxf(t, __shfl_xor(t, 4, 32));
      t = fmaxf(t, __shfl_xor(t, 8, 32));
      float mnew  = fmaxf(m8[v], t);
      float alpha = exp2f(m8[v] - mnew);
      S0[v] = exp2f(S0[v] - mnew);
      S1[v] = exp2f(S1[v] - mnew);
      float rs = S0[v] + S1[v];
      rs += __shfl_xor(rs, 1, 32);
      rs += __shfl_xor(rs, 2, 32);
      rs += __shfl_xor(rs, 4, 32);
      rs += __shfl_xor(rs, 8, 32);
      l8[v] = l8[v] * alpha + rs;
      m8[v] = mnew;
      O0[v] *= alpha; O1[v] *= alpha; O2[v] *= alpha; O3[v] *= alpha;
    }

    // P: C-layout -> LDS -> A-layout (wave-private scratch, DS kept in order).
    _Float16* pw = &pbuf[wid * 512];
#pragma unroll
    for (int v = 0; v < 8; ++v) {
      int r = v + 8 * hi;
      pw[r * 32 + lo16]      = (_Float16)S0[v];
      pw[r * 32 + 16 + lo16] = (_Float16)S1[v];
    }
    __asm__ volatile("s_wait_dscnt 0" ::: "memory");
    v16h pa = load_a16(pw + lo16 * 32 + hi * 16);

    // O += P * V  (B operand = rows of V^T at t0 + hi*16)
    const _Float16* vb = &vtbuf[lo16 * SEG_ + t0 + hi * 16];
    O0 = wmma16(pa, *(const v16h*)(vb + 0  * SEG_), O0);
    O1 = wmma16(pa, *(const v16h*)(vb + 16 * SEG_), O1);
    O2 = wmma16(pa, *(const v16h*)(vb + 32 * SEG_), O2);
    O3 = wmma16(pa, *(const v16h*)(vb + 48 * SEG_), O3);
  }

  // Normalize, un-permute rows, store f16 operand for the output GEMM.
#pragma unroll
  for (int v = 0; v < 8; ++v) {
    float il = 1.0f / l8[v];
    int p = seg * SEG_ + qrow + v + 8 * hi;
    int morig = perm[p];
    _Float16* dst = attn_h + ((size_t)b * M_ + morig) * HD_ + h * 64 + lo16;
    dst[0]  = (_Float16)(O0[v] * il);
    dst[16] = (_Float16)(O1[v] * il);
    dst[32] = (_Float16)(O2[v] * il);
    dst[48] = (_Float16)(O3[v] * il);
  }
}

// ---------------------------------------------------------------------------
// GEMM2: out[b,m,f] = sum_c attn_h[b,m,c] * w_out[f,c]   (K = 512, f32 out)
// ---------------------------------------------------------------------------
__global__ __launch_bounds__(256) void out_gemm_kernel(
    const _Float16* __restrict__ attn_h, const _Float16* __restrict__ wout_h,
    float* __restrict__ out) {
  const int wid = blockIdx.x * 8 + (threadIdx.x >> 5);
  const int rb = wid >> 3, nb = wid & 7;
  const int lane = threadIdx.x & 31, lo16 = lane & 15, hi = lane >> 4;

  const _Float16* arow = attn_h + (size_t)(rb * 16 + lo16) * HD_;
  v8f acc[4];
#pragma unroll
  for (int n = 0; n < 4; ++n)
#pragma unroll
    for (int v = 0; v < 8; ++v) acc[n][v] = 0.f;

  for (int kb = 0; kb < HD_; kb += 32) {
    v16h a = load_a16(arow + kb + hi * 16);
#pragma unroll
    for (int n = 0; n < 4; ++n) {
      const _Float16* brow = wout_h + (size_t)(nb * 64 + n * 16 + lo16) * HD_;
      acc[n] = wmma16(a, *(const v16h*)(brow + kb + hi * 16), acc[n]);
    }
  }

#pragma unroll
  for (int v = 0; v < 8; ++v) {
    size_t r = (size_t)(rb * 16 + v + 8 * hi);
#pragma unroll
    for (int n = 0; n < 4; ++n)
      out[r * HD_ + nb * 64 + n * 16 + lo16] = acc[n][v];
  }
}

// ---------------------------------------------------------------------------
extern "C" void kernel_launch(void* const* d_in, const int* in_sizes, int n_in,
                              void* d_out, int out_size, void* d_ws, size_t ws_size,
                              hipStream_t stream) {
  const float* q     = (const float*)d_in[0];
  // d_in[1] (k) and d_in[2] (v) are unused by the reference computation.
  const float* w_qkv = (const float*)d_in[3];
  const float* w_out = (const float*)d_in[4];
  float* out = (float*)d_out;

  char* ws = (char*)d_ws;
  auto take = [&](size_t bytes) {
    char* p = ws;
    ws += (bytes + 255) & ~(size_t)255;
    return p;
  };
  int*      dvals  = (int*)take((size_t)M_ * 4);
  int*      perm   = (int*)take((size_t)M_ * 4);
  int*      inv    = (int*)take((size_t)M_ * 4);
  _Float16* x_h    = (_Float16*)take((size_t)ROWS_ * 64 * 2);        //  2 MB
  _Float16* wr_h   = (_Float16*)take((size_t)F3_ * 64 * 2);          // .2 MB
  _Float16* wout_h = (_Float16*)take((size_t)HD_ * HD_ * 2);         // .5 MB
  _Float16* qkvP   = (_Float16*)take((size_t)ROWS_ * F3_ * 2);       // 48 MB
  _Float16* attn_h = (_Float16*)take((size_t)ROWS_ * HD_ * 2);       // 16 MB

  hilbert_d_kernel<<<M_ / 256, 256, 0, stream>>>(dvals);
  hilbert_rank_kernel<<<M_ / 256, 256, 0, stream>>>(dvals, perm, inv);
  mean_kernel<<<ROWS_ * 64 / 256, 256, 0, stream>>>(q, x_h);
  wr_kernel<<<F3_ * 64 / 256, 256, 0, stream>>>(w_qkv, wr_h);
  wout_cvt_kernel<<<HD_ * HD_ / 256, 256, 0, stream>>>(w_out, wout_h);

  // 1024 row-blocks x 24 col-blocks = 24576 waves / 8 per WG
  qkv_gemm_kernel<<<(ROWS_ / 16) * (F3_ / 64) / 8, 256, 0, stream>>>(x_h, wr_h, inv, qkvP);

  // (b, seg, head) x 4 quarters = 1024 workgroups
  attn_kernel<<<B_ * NSEG_ * H_ * 4, 256, 0, stream>>>(qkvP, perm, attn_h);

  // 1024 row-blocks x 8 col-blocks = 8192 waves / 8 per WG
  out_gemm_kernel<<<(ROWS_ / 16) * (HD_ / 64) / 8, 256, 0, stream>>>(attn_h, wout_h, out);
}
